// EGRUBlock_59304908423715
// MI455X (gfx1250) — matrix operand order repaired and checked by reading, not compiled, not run on hardware
//
#include <hip/hip_runtime.h>
#include <hip/hip_bf16.h>
#include <stdint.h>

#define B_ 32
#define T_ 2048
#define D_ 1024
#define H_ 1024
#define EPS_ 1e-5f

typedef __attribute__((ext_vector_type(16))) __bf16 v16bf;
typedef __attribute__((ext_vector_type(8)))  float  v8f;
typedef __attribute__((ext_vector_type(4)))  unsigned int v4u;

union Frag16 { v16bf bf; v4u u[2]; };

// Padded LDS row pitch (in bf16 elements): 40*2 = 80B -> 20-bank stride,
// gcd(20,64)=4 -> 16 consecutive rows hit 16 distinct banks for b128 reads.
#define LDS_PITCH 40

#if defined(__has_builtin)
#if __has_builtin(__builtin_amdgcn_global_load_async_to_lds_b128)
#define HAVE_ASYNC_LDS 1
#endif
#endif
#ifndef HAVE_ASYNC_LDS
#define HAVE_ASYNC_LDS 0
#endif

#if HAVE_ASYNC_LDS
// Exact parameter types per hipcc diagnostic:
//   __attribute__((__vector_size__(4*sizeof(int)))) int  in global / LDS AS
typedef int v4i_gen __attribute__((vector_size(16)));
typedef __attribute__((address_space(1))) v4i_gen* async_gptr;
typedef __attribute__((address_space(3))) v4i_gen* async_lptr;
#endif

static __device__ inline unsigned short f32_to_bf16(float f) {
  union { float f; uint32_t u; } v; v.f = f;
  const uint32_t r = v.u + 0x7FFFu + ((v.u >> 16) & 1u);   // round-to-nearest-even
  return (unsigned short)(r >> 16);
}

// ---------------------------------------------------------------------------
// WMMA bf16 16x16x32, fp32 accumulate
// ---------------------------------------------------------------------------
static __device__ inline v8f wmma_bf16(v16bf a, v16bf b, v8f c) {
  return __builtin_amdgcn_wmma_f32_16x16x32_bf16(
      /*neg_a=*/false, a, /*neg_b=*/false, b,
      /*c_mod=*/(short)0, c, /*reuse_a=*/false, /*reuse_b=*/false);
}

// A fragment: 16x32 bf16 tile of row-major [M,K] src (pitch ld elements).
// ISA layout: lanes 0-15 row m=lane, halves 0-7 -> K 0-7, halves 8-15 -> K 16-23;
//             lanes 16-31 row m=lane-16, K ranges shifted by +8.
static __device__ inline v16bf load_frag_a(const unsigned short* __restrict__ src,
                                           int row0, int col0, int ld) {
  const int lane = threadIdx.x & 31;
  const int m = lane & 15;
  const int kbase = col0 + ((lane >> 4) << 3);
  const unsigned short* p = src + (size_t)(row0 + m) * ld + kbase;
  Frag16 f;
  f.u[0] = *(const v4u*)(p);        // K = kbase .. kbase+7
  f.u[1] = *(const v4u*)(p + 16);   // K = kbase+16 .. kbase+23
  return f.bf;
}

// B fragment: 32x16 bf16, B[k,n] = W[n,k] with W row-major [N,K] (pitch ld).
// lane holds column n=lane&15; lanes 0-15 K 0-7,16-23, lanes 16-31 K 8-15,24-31
// => two 16-byte runs per lane (matches A fragment).
static __device__ inline v16bf load_frag_b(const unsigned short* __restrict__ w,
                                           int n0, int k0, int ld) {
  const int lane = threadIdx.x & 31;
  const unsigned short* p = w + (size_t)(n0 + (lane & 15)) * ld
                              + (k0 + ((lane >> 4) << 3));
  Frag16 f;
  f.u[0] = *(const v4u*)(p);
  f.u[1] = *(const v4u*)(p + 16);
  return f.bf;
}

// ---------------------------------------------------------------------------
// Async global -> LDS 16-byte copy (CDNA5 GLOBAL_LOAD_ASYNC_TO_LDS_B128).
// Falls back to a plain LDS store if the builtin is unavailable.
// ---------------------------------------------------------------------------
static __device__ inline void async_copy16(const unsigned short* __restrict__ g,
                                           unsigned short* l) {
#if HAVE_ASYNC_LDS
  __builtin_amdgcn_global_load_async_to_lds_b128(
      (async_gptr)(uintptr_t)g,
      (async_lptr)(uint32_t)(uintptr_t)l,   // generic LDS ptr low 32 bits = LDS addr
      0, 0);
#else
  *(v4u*)l = *(const v4u*)g;
#endif
}

static __device__ inline void wait_async() {
#if HAVE_ASYNC_LDS
#if __has_builtin(__builtin_amdgcn_s_wait_asynccnt)
  __builtin_amdgcn_s_wait_asynccnt(0);
#else
  asm volatile("s_wait_asynccnt 0x0" ::: "memory");
#endif
#endif
}

// ---------------------------------------------------------------------------
// LayerNorm over D, fp32 in -> bf16 out. One 256-thread block per row.
// ---------------------------------------------------------------------------
__global__ __launch_bounds__(256) void layernorm_k(const float* __restrict__ x,
                                                   const float* __restrict__ gamma,
                                                   const float* __restrict__ beta,
                                                   unsigned short* __restrict__ xn) {
  __shared__ float red[256];
  const int tid = threadIdx.x;
  const size_t row = blockIdx.x;
  const float* xr = x + row * D_;
  float v[4];
  float s = 0.f;
#pragma unroll
  for (int i = 0; i < 4; ++i) { v[i] = xr[tid + i * 256]; s += v[i]; }
  red[tid] = s; __syncthreads();
  for (int k = 128; k > 0; k >>= 1) { if (tid < k) red[tid] += red[tid + k]; __syncthreads(); }
  const float mu = red[0] * (1.f / (float)D_);
  __syncthreads();
  float vs = 0.f;
#pragma unroll
  for (int i = 0; i < 4; ++i) { const float d = v[i] - mu; vs += d * d; }
  red[tid] = vs; __syncthreads();
  for (int k = 128; k > 0; k >>= 1) { if (tid < k) red[tid] += red[tid + k]; __syncthreads(); }
  const float inv = rsqrtf(red[0] * (1.f / (float)D_) + EPS_);
  unsigned short* xo = xn + row * D_;
#pragma unroll
  for (int i = 0; i < 4; ++i) {
    const int c = tid + i * 256;
    xo[c] = f32_to_bf16((v[i] - mu) * inv * gamma[c] + beta[c]);
  }
}

// ---------------------------------------------------------------------------
// fp32 -> bf16 conversion (weights); zero-init of h
// ---------------------------------------------------------------------------
__global__ __launch_bounds__(256) void cvt_bf16(const float* __restrict__ s,
                                                unsigned short* __restrict__ d, int n) {
  const int i = blockIdx.x * 256 + threadIdx.x;
  if (i < n) d[i] = f32_to_bf16(s[i]);
}

__global__ __launch_bounds__(256) void init_h(float* __restrict__ h,
                                              unsigned short* __restrict__ hbf, int n) {
  const int i = blockIdx.x * 256 + threadIdx.x;
  if (i < n) { h[i] = 0.f; hbf[i] = 0; }
}

// ---------------------------------------------------------------------------
// Big GEMM: out[M,N] = xn[M,K] @ W[N,K]^T + bias.  M=65536, N=K=1024.
// Block = 8 waves, block tile 64x256, wave tile 32x64 (2x4 C tiles).
// A/B tiles double-buffered in LDS via async global->LDS b128 copies.
// ---------------------------------------------------------------------------
static __device__ inline void stage_tiles(const unsigned short* __restrict__ A,
                                          const unsigned short* __restrict__ W,
                                          int mBase, int nBase, int k0,
                                          unsigned short* a_sh,
                                          unsigned short* b_sh) {
  const int t = threadIdx.x;
  {
    // A tile: 64 rows x 32 cols = 256 x 16B chunks, 1 per thread
    const int row = t >> 2, cc = (t & 3) << 3;
    async_copy16(A + (size_t)(mBase + row) * D_ + (k0 + cc),
                 a_sh + row * LDS_PITCH + cc);
  }
#pragma unroll
  for (int j = 0; j < 4; ++j) {
    // B tile: 256 rows x 32 cols = 1024 x 16B chunks, 4 per thread
    const int c = t + j * 256;
    const int row = c >> 2, cc = (c & 3) << 3;
    async_copy16(W + (size_t)(nBase + row) * D_ + (k0 + cc),
                 b_sh + row * LDS_PITCH + cc);
  }
}

__global__ __launch_bounds__(256) void gemm_xnW(const unsigned short* __restrict__ A,
                                                const unsigned short* __restrict__ W,
                                                const float* __restrict__ bias,
                                                float* __restrict__ out) {
  __shared__ unsigned short a_sh[2][64 * LDS_PITCH];
  __shared__ unsigned short b_sh[2][256 * LDS_PITCH];

  const int wave = threadIdx.x >> 5;
  const int mBase = blockIdx.y * 64;
  const int nBase = blockIdx.x * 256;
  const int mloc = (wave >> 2) * 32;   // 0 / 32 within block tile
  const int nloc = (wave & 3) * 64;    // 0..192 within block tile

  v8f acc[2][4] = {};

  stage_tiles(A, W, mBase, nBase, 0, a_sh[0], b_sh[0]);
  wait_async();
  __syncthreads();

  const int NK = D_ / 32;
  for (int kk = 0; kk < NK; ++kk) {
    const int cur = kk & 1;
    if (kk + 1 < NK)
      stage_tiles(A, W, mBase, nBase, (kk + 1) * 32, a_sh[cur ^ 1], b_sh[cur ^ 1]);

    const v16bf a0 = load_frag_a(a_sh[cur], mloc,      0, LDS_PITCH);
    const v16bf a1 = load_frag_a(a_sh[cur], mloc + 16, 0, LDS_PITCH);
#pragma unroll
    for (int j = 0; j < 4; ++j) {
      const v16bf b = load_frag_b(b_sh[cur], nloc + j * 16, 0, LDS_PITCH);
      acc[0][j] = wmma_bf16(a0, b, acc[0][j]);
      acc[1][j] = wmma_bf16(a1, b, acc[1][j]);
    }

    if (kk + 1 < NK) { wait_async(); __syncthreads(); }
  }

  const int lane = threadIdx.x & 31;
  const int ncol = lane & 15;
  const int mofs = (lane >> 4) << 3;
#pragma unroll
  for (int i = 0; i < 2; ++i)
#pragma unroll
    for (int j = 0; j < 4; ++j) {
      const int n = nBase + nloc + j * 16 + ncol;
      const float bn = bias[n];
      float* o = out + (size_t)(mBase + mloc + i * 16 + mofs) * H_ + n;
#pragma unroll
      for (int r = 0; r < 8; ++r) o[(size_t)r * H_] = acc[i][j][r] + bn;
    }
}

// ---------------------------------------------------------------------------
// Recurrent step, phase A: z = sigmoid(az_t + h@Uz^T); rh = sigmoid(ar_t + h@Ur^T)*h
// One block (4 waves, 128 thr) per 16x16 tile; K=1024 split 4 ways (8 wmma/wave),
// partials reduced through LDS -> ~4x shorter dependent chain per step.
// Tiles: gate(2) x mtile(2) x ntile(64) = 256 blocks.
// ---------------------------------------------------------------------------
__global__ __launch_bounds__(128) void step_zr(const unsigned short* __restrict__ hbf,
                                               const unsigned short* __restrict__ Uzb,
                                               const unsigned short* __restrict__ Urb,
                                               const float* __restrict__ az,
                                               const float* __restrict__ ar,
                                               const float* __restrict__ hq,
                                               float* __restrict__ zout,
                                               unsigned short* __restrict__ rhout,
                                               int t) {
  __shared__ float part[4][256];
  const int tile = blockIdx.x;
  const int gate = tile >> 7;
  const int m0   = ((tile >> 6) & 1) * 16;
  const int n0   = (tile & 63) * 16;
  const int w    = threadIdx.x >> 5;
  const int lane = threadIdx.x & 31;
  const unsigned short* U = gate ? Urb : Uzb;

  v8f acc = {};
  const int kBeg = w * 256;
#pragma unroll
  for (int k0 = 0; k0 < 256; k0 += 32)
    acc = wmma_bf16(load_frag_a(hbf, m0, kBeg + k0, H_),
                    load_frag_b(U,   n0, kBeg + k0, H_), acc);

#pragma unroll
  for (int r = 0; r < 8; ++r) part[w][lane * 8 + r] = acc[r];
  __syncthreads();

  if (w == 0) {
    const int n  = n0 + (lane & 15);
    const int mb = m0 + ((lane >> 4) << 3);
    if (gate == 0) {
#pragma unroll
      for (int r = 0; r < 8; ++r) {
        const int m = mb + r;
        const float s = part[0][lane * 8 + r] + part[1][lane * 8 + r] +
                        part[2][lane * 8 + r] + part[3][lane * 8 + r];
        const float pre = s + az[((size_t)m * T_ + t) * H_ + n];
        zout[m * H_ + n] = 1.f / (1.f + __expf(-pre));
      }
    } else {
#pragma unroll
      for (int r = 0; r < 8; ++r) {
        const int m = mb + r;
        const float s = part[0][lane * 8 + r] + part[1][lane * 8 + r] +
                        part[2][lane * 8 + r] + part[3][lane * 8 + r];
        const float pre = s + ar[((size_t)m * T_ + t) * H_ + n];
        const float rg = 1.f / (1.f + __expf(-pre));
        rhout[m * H_ + n] = f32_to_bf16(rg * hq[m * H_ + n]);
      }
    }
  }
}

// ---------------------------------------------------------------------------
// Recurrent step, phase B: h_tilde = tanh(ah_t + rh@Uh^T);
// h = (1-z)h + z*h_tilde;  y[t] = h + x[t].  128 blocks (2 mtile x 64 ntile).
// ---------------------------------------------------------------------------
__global__ __launch_bounds__(128) void step_h(const unsigned short* __restrict__ rhbf,
                                              const unsigned short* __restrict__ Uhb,
                                              const float* __restrict__ ah,
                                              const float* __restrict__ zbuf,
                                              const float* __restrict__ x,
                                              float* __restrict__ h,
                                              unsigned short* __restrict__ hbf,
                                              float* __restrict__ out,
                                              int t) {
  __shared__ float part[4][256];
  const int tile = blockIdx.x;
  const int m0   = (tile >> 6) * 16;
  const int n0   = (tile & 63) * 16;
  const int w    = threadIdx.x >> 5;
  const int lane = threadIdx.x & 31;

  v8f acc = {};
  const int kBeg = w * 256;
#pragma unroll
  for (int k0 = 0; k0 < 256; k0 += 32)
    acc = wmma_bf16(load_frag_a(rhbf, m0, kBeg + k0, H_),
                    load_frag_b(Uhb,  n0, kBeg + k0, H_), acc);

#pragma unroll
  for (int r = 0; r < 8; ++r) part[w][lane * 8 + r] = acc[r];
  __syncthreads();

  if (w == 0) {
    const int n  = n0 + (lane & 15);
    const int mb = m0 + ((lane >> 4) << 3);
#pragma unroll
    for (int r = 0; r < 8; ++r) {
      const int m = mb + r;
      const float s = part[0][lane * 8 + r] + part[1][lane * 8 + r] +
                      part[2][lane * 8 + r] + part[3][lane * 8 + r];
      const size_t seq = ((size_t)m * T_ + t) * H_ + n;
      const float ht = tanhf(s + ah[seq]);
      const float zz = zbuf[m * H_ + n];
      const float hn = (1.f - zz) * h[m * H_ + n] + zz * ht;
      h[m * H_ + n] = hn;
      hbf[m * H_ + n] = f32_to_bf16(hn);
      out[seq] = hn + x[seq];   // residual (D == H)
    }
  }
}

// ---------------------------------------------------------------------------
// Host-side orchestration
// ---------------------------------------------------------------------------
extern "C" void kernel_launch(void* const* d_in, const int* in_sizes, int n_in,
                              void* d_out, int out_size, void* d_ws, size_t ws_size,
                              hipStream_t stream) {
  const float* x     = (const float*)d_in[0];
  const float* Wz    = (const float*)d_in[1];
  const float* bz    = (const float*)d_in[2];
  const float* Uz    = (const float*)d_in[3];
  const float* Wr    = (const float*)d_in[4];
  const float* br    = (const float*)d_in[5];
  const float* Ur    = (const float*)d_in[6];
  const float* Wh    = (const float*)d_in[7];
  const float* bh    = (const float*)d_in[8];
  const float* Uh    = (const float*)d_in[9];
  const float* gamma = (const float*)d_in[10];
  const float* beta  = (const float*)d_in[11];
  float* out = (float*)d_out;

  uint8_t* ws = (uint8_t*)d_ws;
  size_t off = 0;
  auto take = [&](size_t bytes) -> void* {
    void* p = ws + off;
    off += (bytes + 255) & ~(size_t)255;
    return p;
  };
  const size_t BT = (size_t)B_ * T_;

  unsigned short* xnb  = (unsigned short*)take(BT * D_ * 2);
  unsigned short* Wzb  = (unsigned short*)take((size_t)H_ * D_ * 2);
  unsigned short* Wrb  = (unsigned short*)take((size_t)H_ * D_ * 2);
  unsigned short* Whb  = (unsigned short*)take((size_t)H_ * D_ * 2);
  unsigned short* Uzb  = (unsigned short*)take((size_t)H_ * H_ * 2);
  unsigned short* Urb  = (unsigned short*)take((size_t)H_ * H_ * 2);
  unsigned short* Uhb  = (unsigned short*)take((size_t)H_ * H_ * 2);
  float*          az   = (float*)take(BT * H_ * 4);
  float*          ar   = (float*)take(BT * H_ * 4);
  float*          ah   = (float*)take(BT * H_ * 4);
  float*          h    = (float*)take((size_t)B_ * H_ * 4);
  unsigned short* hbf  = (unsigned short*)take((size_t)B_ * H_ * 2);
  float*          zbuf = (float*)take((size_t)B_ * H_ * 4);
  unsigned short* rhbf = (unsigned short*)take((size_t)B_ * H_ * 2);

  // 1) LayerNorm -> bf16 activations
  layernorm_k<<<(unsigned)BT, 256, 0, stream>>>(x, gamma, beta, xnb);

  // 2) Weights -> bf16
  const int NW = H_ * D_;
  cvt_bf16<<<NW / 256, 256, 0, stream>>>(Wz, Wzb, NW);
  cvt_bf16<<<NW / 256, 256, 0, stream>>>(Wr, Wrb, NW);
  cvt_bf16<<<NW / 256, 256, 0, stream>>>(Wh, Whb, NW);
  cvt_bf16<<<NW / 256, 256, 0, stream>>>(Uz, Uzb, NW);
  cvt_bf16<<<NW / 256, 256, 0, stream>>>(Ur, Urb, NW);
  cvt_bf16<<<NW / 256, 256, 0, stream>>>(Uh, Uhb, NW);

  // 3) Parallel input-side GEMMs (az/ar/ah), WMMA bf16 + async LDS staging
  dim3 gg(H_ / 256, (unsigned)(BT / 64));
  gemm_xnW<<<gg, 256, 0, stream>>>(xnb, Wzb, bz, az);
  gemm_xnW<<<gg, 256, 0, stream>>>(xnb, Wrb, br, ar);
  gemm_xnW<<<gg, 256, 0, stream>>>(xnb, Whb, bh, ah);

  // 4) h0 = 0
  init_h<<<(B_ * H_) / 256, 256, 0, stream>>>(h, hbf, B_ * H_);

  // 5) Sequential scan: 2 kernels per timestep (r must be final before Uh GEMM)
  for (int t = 0; t < T_; ++t) {
    step_zr<<<256, 128, 0, stream>>>(hbf, Uzb, Urb, az, ar, h, zbuf, rhbf, t);
    step_h<<<128, 128, 0, stream>>>(rhbf, Uhb, ah, zbuf, x, h, hbf, out, t);
  }
}
